// PoseGCN_72267119722647
// MI455X (gfx1250) — matrix operand order repaired and checked by reading, not compiled, not run on hardware
//
#include <hip/hip_runtime.h>

#define NN 139264      // nodes
#define NE 262144      // edges
#define NG 8192        // graphs
#define NPG 17         // nodes per graph

typedef __attribute__((ext_vector_type(16))) _Float16 v16h;
typedef __attribute__((ext_vector_type(8)))  _Float16 v8h;
typedef __attribute__((ext_vector_type(8)))  float    v8f;

#define CDIV(a,b) (((a)+(b)-1)/(b))

// ---------------- small utility kernels ----------------

__global__ void k_zero_i32(int* __restrict__ p, int n) {
  int i = blockIdx.x * blockDim.x + threadIdx.x;
  if (i < n) p[i] = 0;
}

__global__ void k_degree(const int* __restrict__ dst, int* __restrict__ deg, int e) {
  int i = blockIdx.x * blockDim.x + threadIdx.x;
  if (i < e) atomicAdd(&deg[dst[i]], 1);
}

__global__ void k_dinv(const int* __restrict__ deg, float* __restrict__ dinv, int n) {
  int i = blockIdx.x * blockDim.x + threadIdx.x;
  if (i < n) dinv[i] = rsqrtf((float)deg[i] + 1.0f);   // +1 for self-loop
}

__global__ void k_norm(const int* __restrict__ src, const int* __restrict__ dst,
                       const float* __restrict__ dinv, float* __restrict__ nrm, int e) {
  int i = blockIdx.x * blockDim.x + threadIdx.x;
  if (i < e) nrm[i] = dinv[src[i]] * dinv[dst[i]];
}

// Pack W (K x Nc row-major, f32) into the WMMA B-operand per-lane layout (f16).
// Packed element idx = ((kt*ntiles + nt)*32 + lane)*16 + e  holds
// W[kt*32 + (lane>>4)*8 + (e&7) + (e>>3)*16][nt*16 + (lane&15)].
__global__ void k_packB(const float* __restrict__ W, _Float16* __restrict__ Bp,
                        int K, int Nc) {
  int idx = blockIdx.x * blockDim.x + threadIdx.x;
  if (idx >= K * Nc) return;
  int e    = idx & 15;
  int lane = (idx >> 4) & 31;
  int tile = idx >> 9;                 // / (32*16)
  int ntiles = Nc >> 4;
  int kt = tile / ntiles, nt = tile % ntiles;
  int k = kt * 32 + (lane >> 4) * 8 + (e & 7) + (e >> 3) * 16;
  int n = nt * 16 + (lane & 15);
  Bp[idx] = (_Float16)W[k * Nc + n];
}

// Layer 1 linear: t = x @ W1  (x: N x 3, W1: 3 x 64) -- K=3 too small for WMMA.
__global__ void k_l1(const float* __restrict__ x, const float* __restrict__ W1,
                     float* __restrict__ t, int n) {
  int idx = blockIdx.x * blockDim.x + threadIdx.x;
  if (idx >= n * 64) return;
  int i = idx >> 6, f = idx & 63;
  float x0 = x[i * 3 + 0], x1 = x[i * 3 + 1], x2 = x[i * 3 + 2];
  t[idx] = x0 * W1[f] + x1 * W1[64 + f] + x2 * W1[128 + f];
}

// agg = t * dinv^2 (self-loop contribution, initializes the scatter target)
__global__ void k_agg_init(const float* __restrict__ t, const float* __restrict__ dinv,
                           float* __restrict__ agg, int n, int F) {
  int idx = blockIdx.x * blockDim.x + threadIdx.x;
  if (idx >= n * F) return;
  float d = dinv[idx / F];
  agg[idx] = t[idx] * d * d;
}

// agg[dst] += norm * t[src]   (hardware f32 atomic add in L2)
__global__ void k_agg_edges(const int* __restrict__ src, const int* __restrict__ dst,
                            const float* __restrict__ nrm, const float* __restrict__ t,
                            float* __restrict__ agg, int e, int F) {
  int idx = blockIdx.x * blockDim.x + threadIdx.x;
  if (idx >= e * F) return;
  int ed = idx / F, f = idx - ed * F;
  unsafeAtomicAdd(&agg[dst[ed] * F + f], nrm[ed] * t[src[ed] * F + f]);
}

// out = relu(agg + bias); optionally emit f16 (next WMMA A) and/or f32
__global__ void k_finish(const float* __restrict__ agg, const float* __restrict__ bias,
                         _Float16* __restrict__ outh, float* __restrict__ outf,
                         int n, int F) {
  int idx = blockIdx.x * blockDim.x + threadIdx.x;
  if (idx >= n * F) return;
  float v = fmaxf(agg[idx] + bias[idx % F], 0.0f);
  if (outh) outh[idx] = (_Float16)v;
  if (outf) outf[idx] = v;
}

// ---------------- WMMA GEMM: C[M x Nc] = A[M x K] * Bpacked ----------------
// One wave per 16x16 output tile. M%16==0, Nc%16==0, K%32==0.
// Epilogue: optional bias add (per column), optional relu; writes f32 and/or f16.
__global__ void __launch_bounds__(256)
k_gemm_wmma(const _Float16* __restrict__ A, const _Float16* __restrict__ Bp,
            const float* __restrict__ bias, float* __restrict__ Cf,
            _Float16* __restrict__ Ch, int M, int Nc, int K, int relu) {
  const int lane = threadIdx.x & 31;
  const int wave = threadIdx.x >> 5;
  const int ntiles = Nc >> 4;
  const int mtiles = M >> 4;
  int tile = blockIdx.x * (blockDim.x >> 5) + wave;   // uniform per wave
  if (tile >= mtiles * ntiles) return;                // whole-wave exit: EXEC stays all-1s
  const int tm = tile / ntiles, tn = tile % ntiles;
  const int r = lane & 15, h = lane >> 4;

  const _Float16* Arow = A + (size_t)(tm * 16 + r) * K;
  v8f acc = {};
  const int ktiles = K >> 5;
  for (int kt = 0; kt < ktiles; ++kt) {
    const int k0 = kt * 32;
    // A operand: lane holds row M=r, K = k0+h*8..+7 and k0+16+h*8..+7
    v8h alo = *(const v8h*)(Arow + k0 + h * 8);
    v8h ahi = *(const v8h*)(Arow + k0 + 16 + h * 8);
    v16h a;
#pragma unroll
    for (int i = 0; i < 8; ++i) { a[i] = alo[i]; a[8 + i] = ahi[i]; }
    // B operand: pre-packed, one aligned 32B chunk per lane
    v16h b = *(const v16h*)(Bp + ((size_t)(kt * ntiles + tn) * 32 + lane) * 16);
    acc = __builtin_amdgcn_wmma_f32_16x16x32_f16(false, a, false, b,
                                                 (short)0, acc, false, false);
  }
#pragma unroll
  for (int v = 0; v < 8; ++v) {
    int orow = tm * 16 + v + 8 * h;      // C/D layout: VGPR v, half h -> M = v + 8h
    int ocol = tn * 16 + r;              // N = lane & 15
    float val = acc[v];
    if (bias) val += bias[ocol];
    if (relu) val = fmaxf(val, 0.0f);
    size_t o = (size_t)orow * Nc + ocol;
    if (Cf) Cf[o] = val;
    if (Ch) Ch[o] = (_Float16)val;
  }
}

// global mean pool over the 17 nodes of each graph (counts are exactly 17)
__global__ void k_pool(const float* __restrict__ h3, float* __restrict__ pf,
                       _Float16* __restrict__ ph, int g, int F) {
  int idx = blockIdx.x * blockDim.x + threadIdx.x;
  if (idx >= g * F) return;
  int gr = idx / F, f = idx - gr * F;
  float s = 0.0f;
  const float* base = h3 + (size_t)gr * NPG * F + f;
#pragma unroll
  for (int j = 0; j < NPG; ++j) s += base[(size_t)j * F];
  float m = s * (1.0f / (float)NPG);
  pf[idx] = m;
  ph[idx] = (_Float16)m;
}

// final 64->2 linear + log_softmax (per graph)
__global__ void k_head(const _Float16* __restrict__ z2, const float* __restrict__ Wf3,
                       const float* __restrict__ bf3, float* __restrict__ out, int g) {
  int i = blockIdx.x * blockDim.x + threadIdx.x;
  if (i >= g) return;
  float z0 = bf3[0], z1 = bf3[1];
  const _Float16* zr = z2 + (size_t)i * 64;
#pragma unroll
  for (int k = 0; k < 64; ++k) {
    float a = (float)zr[k];
    z0 += a * Wf3[k * 2 + 0];
    z1 += a * Wf3[k * 2 + 1];
  }
  float m = fmaxf(z0, z1);
  float ls = m + logf(expf(z0 - m) + expf(z1 - m));
  out[i * 2 + 0] = z0 - ls;
  out[i * 2 + 1] = z1 - ls;
}

// ---------------- host-side launch ----------------

extern "C" void kernel_launch(void* const* d_in, const int* in_sizes, int n_in,
                              void* d_out, int out_size, void* d_ws, size_t ws_size,
                              hipStream_t stream) {
  const float* x   = (const float*)d_in[0];
  const int*   src = (const int*)d_in[1];
  const int*   dst = (const int*)d_in[2];
  // d_in[3] = batch (implicit: node i -> graph i/17)
  const float* W1  = (const float*)d_in[4];
  const float* b1  = (const float*)d_in[5];
  const float* W2  = (const float*)d_in[6];
  const float* b2  = (const float*)d_in[7];
  const float* W3  = (const float*)d_in[8];
  const float* b3  = (const float*)d_in[9];
  const float* Wf1 = (const float*)d_in[10];
  const float* bf1 = (const float*)d_in[11];
  const float* Wf2 = (const float*)d_in[12];
  const float* bf2 = (const float*)d_in[13];
  const float* Wf3 = (const float*)d_in[14];
  const float* bf3 = (const float*)d_in[15];
  float* out = (float*)d_out;

  // workspace carving (256B aligned); T/AGG regions reused across all 3 GCN layers
  char* w = (char*)d_ws;
  auto carve = [&](size_t bytes) -> void* {
    void* p = (void*)w;
    w += (bytes + 255) & ~(size_t)255;
    return p;
  };
  int*      deg  = (int*)carve((size_t)NN * 4);
  float*    dinv = (float*)carve((size_t)NN * 4);
  float*    nrm  = (float*)carve((size_t)NE * 4);
  float*    T    = (float*)carve((size_t)NN * 256 * 4);   // t1/t2/t3
  float*    AGG  = (float*)carve((size_t)NN * 256 * 4);   // agg1/agg2/agg3 (+h3 f32 in place)
  _Float16* h1h  = (_Float16*)carve((size_t)NN * 64 * 2);
  _Float16* h2h  = (_Float16*)carve((size_t)NN * 128 * 2);
  _Float16* W2p  = (_Float16*)carve((size_t)64 * 128 * 2);
  _Float16* W3p  = (_Float16*)carve((size_t)128 * 256 * 2);
  _Float16* Wf1p = (_Float16*)carve((size_t)256 * 128 * 2);
  _Float16* Wf2p = (_Float16*)carve((size_t)128 * 64 * 2);
  float*    pf   = (float*)carve((size_t)NG * 256 * 4);
  _Float16* ph   = (_Float16*)carve((size_t)NG * 256 * 2);
  _Float16* z1h  = (_Float16*)carve((size_t)NG * 128 * 2);
  _Float16* z2h  = (_Float16*)carve((size_t)NG * 64 * 2);

  const int B = 256;

  // degrees + symmetric norm coefficients
  k_zero_i32<<<CDIV(NN, B), B, 0, stream>>>(deg, NN);
  k_degree<<<CDIV(NE, B), B, 0, stream>>>(dst, deg, NE);
  k_dinv<<<CDIV(NN, B), B, 0, stream>>>(deg, dinv, NN);
  k_norm<<<CDIV(NE, B), B, 0, stream>>>(src, dst, dinv, nrm, NE);

  // pack weights into WMMA B layout
  k_packB<<<CDIV(64 * 128, B), B, 0, stream>>>(W2, W2p, 64, 128);
  k_packB<<<CDIV(128 * 256, B), B, 0, stream>>>(W3, W3p, 128, 256);
  k_packB<<<CDIV(256 * 128, B), B, 0, stream>>>(Wf1, Wf1p, 256, 128);
  k_packB<<<CDIV(128 * 64, B), B, 0, stream>>>(Wf2, Wf2p, 128, 64);

  // ---- GCN layer 1 (3 -> 64), scalar matmul ----
  k_l1<<<CDIV(NN * 64, B), B, 0, stream>>>(x, W1, T, NN);
  k_agg_init<<<CDIV(NN * 64, B), B, 0, stream>>>(T, dinv, AGG, NN, 64);
  k_agg_edges<<<CDIV(NE * 64, B), B, 0, stream>>>(src, dst, nrm, T, AGG, NE, 64);
  k_finish<<<CDIV(NN * 64, B), B, 0, stream>>>(AGG, b1, h1h, nullptr, NN, 64);

  // ---- GCN layer 2 (64 -> 128), WMMA ----
  {
    int tiles = (NN / 16) * (128 / 16);
    k_gemm_wmma<<<CDIV(tiles, 8), B, 0, stream>>>(h1h, W2p, nullptr, T, nullptr,
                                                  NN, 128, 64, 0);
  }
  k_agg_init<<<CDIV(NN * 128, B), B, 0, stream>>>(T, dinv, AGG, NN, 128);
  k_agg_edges<<<CDIV(NE * 128, B), B, 0, stream>>>(src, dst, nrm, T, AGG, NE, 128);
  k_finish<<<CDIV(NN * 128, B), B, 0, stream>>>(AGG, b2, h2h, nullptr, NN, 128);

  // ---- GCN layer 3 (128 -> 256), WMMA ----
  {
    int tiles = (NN / 16) * (256 / 16);
    k_gemm_wmma<<<CDIV(tiles, 8), B, 0, stream>>>(h2h, W3p, nullptr, T, nullptr,
                                                  NN, 256, 128, 0);
  }
  k_agg_init<<<CDIV(NN * 256, B), B, 0, stream>>>(T, dinv, AGG, NN, 256);
  k_agg_edges<<<CDIV(NE * 256, B), B, 0, stream>>>(src, dst, nrm, T, AGG, NE, 256);
  k_finish<<<CDIV(NN * 256, B), B, 0, stream>>>(AGG, b3, nullptr, AGG, NN, 256); // h3 f32 in place

  // ---- mean pool + MLP head ----
  k_pool<<<CDIV(NG * 256, B), B, 0, stream>>>(AGG, pf, ph, NG, 256);
  {
    int tiles = (NG / 16) * (128 / 16);   // 8192 x 256 @ 256 x 128, bias+relu
    k_gemm_wmma<<<CDIV(tiles, 8), B, 0, stream>>>(ph, Wf1p, bf1, nullptr, z1h,
                                                  NG, 128, 256, 1);
  }
  {
    int tiles = (NG / 16) * (64 / 16);    // 8192 x 128 @ 128 x 64, bias+relu
    k_gemm_wmma<<<CDIV(tiles, 8), B, 0, stream>>>(z1h, Wf2p, bf2, nullptr, z2h,
                                                  NG, 64, 128, 1);
  }
  k_head<<<CDIV(NG, B), B, 0, stream>>>(z2h, Wf3, bf3, out, NG);

  (void)in_sizes; (void)n_in; (void)out_size; (void)ws_size;
}
